// ImprovedGate_86689619902981
// MI455X (gfx1250) — compile-verified, matrix operand order checked by reading, not run on hardware
//
#include <hip/hip_runtime.h>
#include <hip/hip_bf16.h>
#include <math.h>

// CDNA5 WMMA vector types (wave32): A/B = 16 halves/lane, C/D = 8 floats/lane
typedef __attribute__((ext_vector_type(16))) _Float16 v16h;
typedef __attribute__((ext_vector_type(8)))  _Float16 v8h;
typedef __attribute__((ext_vector_type(8)))  float    v8f;
typedef __attribute__((ext_vector_type(4)))  float    v4f;
typedef __attribute__((ext_vector_type(4)))  unsigned int v4u;
typedef __attribute__((ext_vector_type(8)))  int      v8i;
typedef __attribute__((ext_vector_type(4)))  int      v4i;

#define CONV_K 147              // 3*7*7
#define XMAX   (3*224*224 - 1)  // last valid element of one image

// K index held by half h of a lane's A/B fragment (16-bit 16x32 layout):
// lanes 0-15: VGPR0..3 = K0..7, VGPR4..7 = K16..23 ; lanes 16-31: +8
__device__ __forceinline__ int kidx(int h, int lane) {
    return (h < 8 ? h : h + 8) + ((lane & 16) ? 8 : 0);
}

// ---------------------------------------------------------------------------
// Kernel 1: fused conv7x7/s4 (implicit GEMM via WMMA) + bias + ReLU
//           + maxpool3x3/s2 + adaptive-avg-pool(4x4), one (image, row-bin)
//           per block. Conv weights staged into LDS via the Tensor Data
//           Mover. Writes f16 features [B,256] to workspace.
// ---------------------------------------------------------------------------
__global__ __launch_bounds__(256) void conv_pool_kernel(
    const float* __restrict__ x,        // [256,3,224,224]
    const float* __restrict__ conv_w,   // [16,3,7,7] -> flat [16,147]
    const float* __restrict__ conv_b,   // [16]
    _Float16*    __restrict__ feats)    // [256,256] f16
{
    const int bin  = blockIdx.x;        // adaptive-pool row bin 0..3
    const int img  = blockIdx.y;        // image 0..255
    const int tid  = threadIdx.x;
    const int lane = tid & 31;
    const int wave = tid >> 5;          // 8 waves

    // adaptive pool bins over the 27-row maxpool output (PyTorch semantics)
    const int R0[4] = {0, 6, 13, 20};
    const int R1[4] = {7, 14, 21, 27};
    const int r0 = R0[bin], r1 = R1[bin];
    const int rows_mp   = r1 - r0;          // 7 or 8 maxpool rows
    const int conv_row0 = 2 * r0;           // first conv row needed
    const int nrows     = 2 * rows_mp + 1;  // 15 or 17 conv rows
    const int band_px   = nrows * 56;
    const int ntiles    = (band_px + 15) >> 4;

    __shared__ _Float16 conv_s[17 * 56 * 16];  // conv+relu band tile (f16), 30464 B
    __shared__ float    wsh[16 * CONV_K];      // conv weights, 9408 B (TDM dest)
    __shared__ int      tab_s[160];            // packed k -> (koff,ky,kx)
    __shared__ float    pool_s[64];            // [16 ch][4 col bins]

    if (tid < 64) pool_s[tid] = 0.0f;

    // packed decode table: entry = (c*50176 + ky*224 + kx) << 6 | ky<<3 | kx
    if (tid < 160) {
        int k  = tid < CONV_K ? tid : CONV_K - 1;
        int c  = k / 49;
        int rm = k - c * 49;
        int ky = rm / 7;
        int kx = rm - ky * 7;
        tab_s[tid] = ((c * 50176 + ky * 224 + kx) << 6) | (ky << 3) | kx;
    }

    // --- stage conv weights into LDS via the Tensor Data Mover -------------
#if __has_builtin(__builtin_amdgcn_tensor_load_to_lds)
    if (tid < 32) {
        // D# group 0: count=1 | lds_addr | global_addr(57b) | type=2
        unsigned int lds_off = (unsigned int)(uintptr_t)(void*)wsh;
        unsigned long long ga = (unsigned long long)(uintptr_t)conv_w;
        v4u g0;
        g0[0] = 1u;                                        // count=1, user mode
        g0[1] = lds_off;                                   // lds_addr (bytes)
        g0[2] = (unsigned int)(ga & 0xFFFFFFFFu);          // global_addr[31:0]
        g0[3] = (unsigned int)((ga >> 32) & 0x01FFFFFFu)   // global_addr[56:32]
              | (2u << 30);                                // type=2 ("image")
        // D# group 1: 1-D tensor of 2352 f32, tile = whole tensor
        v8i g1 = (v8i)0;
        g1[0] = (int)(2u << 16);                  // data_size = 4 bytes
        g1[1] = (int)((2352u & 0xFFFFu) << 16);   // tensor_dim0[15:0] @bits63:48
        g1[2] = (int)((2352u >> 16) | (1u << 16));// dim0[31:16], tensor_dim1=1
        g1[3] = (int)(2352u << 16);               // tile_dim0 @bits127:112
        g1[4] = 0;                                // tile_dim1/2 = 0 (unused)
        g1[5] = 2352;                             // tensor_dim0_stride[31:0]
        g1[6] = 0;
        g1[7] = 0;
        v4i g2 = (v4i)0;
        v4i g3 = (v4i)0;
        v8i g4 = (v8i)0;                          // extra group (6-arg form)
        __builtin_amdgcn_tensor_load_to_lds(g0, g1, g2, g3, g4, 0);
        __builtin_amdgcn_s_wait_tensorcnt(0);
    }
#else
    for (int i = tid; i < 16 * CONV_K; i += 256) wsh[i] = conv_w[i];
#endif
    __syncthreads();

    // --- B fragments: conv weights as K x N (147x16, K padded to 160) ------
    const int n = lane & 15;                // output channel this lane owns
    v16h bfrag[5];
#pragma unroll
    for (int s = 0; s < 5; ++s) {
#pragma unroll
        for (int h = 0; h < 16; ++h) {
            int k  = s * 32 + kidx(h, lane);
            int kc = k < CONV_K ? k : 0;
            float w = wsh[n * CONV_K + kc];
            bfrag[s][h] = (_Float16)(k < CONV_K ? w : 0.0f);
        }
    }
    const float bias = conv_b[n];
    const float* xb = x + (size_t)img * 3 * 224 * 224;

    // --- implicit-GEMM tiles: 16 output pixels x 16 channels per WMMA ------
    for (int t = wave; t < ntiles; t += 8) {
        const int m   = lane & 15;          // A-matrix row = pixel in tile
        const int pix = t * 16 + m;
        const int oy  = conv_row0 + pix / 56;
        const int ox  = pix % 56;
        const int by  = oy * 4 - 3;         // pad=3, stride=4
        const int bx  = ox * 4 - 3;
        const int blin = by * 224 + bx;

        // prefetch next tile's input (global_prefetch_b8)
        {
            int pn = (t + 8) * 16 + m;
            if (pn < band_px) {
                int oyn = conv_row0 + pn / 56;
                int oxn = pn % 56;
                __builtin_prefetch(xb + (size_t)(oyn * 4) * 224 + oxn * 4, 0, 1);
            }
        }

        v8f acc = {};
#pragma unroll
        for (int s = 0; s < 5; ++s) {
            v16h a;
#pragma unroll
            for (int h = 0; h < 16; ++h) {
                int e    = tab_s[s * 32 + kidx(h, lane)];
                int ky   = (e >> 3) & 7;
                int kx   = e & 7;
                int koff = ((unsigned)e) >> 6;
                int iy   = by + ky;
                int ix   = bx + kx;
                // branch-free: clamp address, zero via multiplicative mask
                int addr = blin + (int)koff;
                addr = addr < 0 ? 0 : (addr > XMAX ? XMAX : addr);
                float msk = ((unsigned)iy < 224u && (unsigned)ix < 224u) ? 1.0f : 0.0f;
                a[h] = (_Float16)(xb[addr] * msk);
            }
            acc = __builtin_amdgcn_wmma_f32_16x16x32_f16(
                false, a, false, bfrag[s], (short)0, acc, false, false);
        }

        // C/D layout: VGPR r, lane: row = r + 8*(lane>=16), col = lane&15
        const int mrow = (lane >> 4) << 3;
#pragma unroll
        for (int r = 0; r < 8; ++r) {
            int p = t * 16 + mrow + r;
            if (p < band_px) {
                float v = acc[r] + bias;
                conv_s[p * 16 + n] = (_Float16)(v > 0.0f ? v : 0.0f);
            }
        }
    }
    __syncthreads();

    // --- maxpool 3x3 s2 + adaptive-avg accumulation ------------------------
    const int C0[4] = {0, 6, 13, 20};
    const int C1[4] = {7, 14, 21, 27};
    const int items = rows_mp * 27 * 16;
    for (int e = tid; e < items; e += 256) {
        int ch  = e & 15;
        int tmp = e >> 4;
        int px  = tmp % 27;
        int pyl = tmp / 27;                  // maxpool row within this bin
        float mx = -INFINITY;
#pragma unroll
        for (int dy = 0; dy < 3; ++dy)
#pragma unroll
            for (int dx = 0; dx < 3; ++dx) {
                float v = (float)conv_s[((2 * pyl + dy) * 56 + (2 * px + dx)) * 16 + ch];
                mx = fmaxf(mx, v);
            }
#pragma unroll
        for (int cb = 0; cb < 4; ++cb) {
            if (px >= C0[cb] && px < C1[cb]) {
                float w = 1.0f / (float)(rows_mp * (C1[cb] - C0[cb]));
                atomicAdd(&pool_s[ch * 4 + cb], mx * w);
            }
        }
    }
    __syncthreads();

    // feats flatten order: ch*16 + ry*4 + rx
    if (tid < 64) {
        int ch = tid >> 2, cb = tid & 3;
        feats[(size_t)img * 256 + ch * 16 + bin * 4 + cb] =
            (_Float16)pool_s[ch * 4 + cb];
    }
}

// ---------------------------------------------------------------------------
// Kernel 2: FC1(256->64)+ReLU and FC2(64->16) via WMMA, temperature,
//           top-2 softmax gates, scatter, all three outputs.
//           16 batch rows per block, 4 waves. Fragment loads are vectorized:
//           each lane's 16 halves are two contiguous 8-element runs.
// ---------------------------------------------------------------------------
__global__ __launch_bounds__(128) void fc_gate_kernel(
    const _Float16* __restrict__ feats,   // [256,256] f16
    const float* __restrict__ fc1_w,      // [64,256]
    const float* __restrict__ fc1_b,      // [64]
    const float* __restrict__ fc2_w,      // [16,64]
    const float* __restrict__ fc2_b,      // [16]
    const float* __restrict__ temp,       // [1]
    float* __restrict__ gates,            // [256,16]
    int*   __restrict__ idx_out,          // [256,2]
    float* __restrict__ logits_out)       // [256,16]
{
    const int tid  = threadIdx.x;
    const int lane = tid & 31;
    const int wave = tid >> 5;            // 4 waves
    const int row0 = blockIdx.x * 16;
    const int kb   = (lane & 16) ? 8 : 0; // lane-half K offset

    __shared__ _Float16 hid_s[16 * 64];
    __shared__ float    logit_s[16 * 16];

    // --- FC1: wave w computes hidden cols [w*16, w*16+16), K=256 -----------
    {
        const int nh = wave * 16 + (lane & 15);
        const _Float16* arow = feats + (size_t)(row0 + (lane & 15)) * 256;
        const float*    wrow = fc1_w + (size_t)nh * 256;
        v8f acc = {};
#pragma unroll
        for (int s = 0; s < 8; ++s) {
            const int k0 = s * 32 + kb;
            // A: halves {k0..k0+7, k0+16..k0+23} -> two 16B loads
            v8h alo = *(const v8h*)(arow + k0);
            v8h ahi = *(const v8h*)(arow + k0 + 16);
            v16h a = __builtin_shufflevector(alo, ahi,
                0, 1, 2, 3, 4, 5, 6, 7, 8, 9, 10, 11, 12, 13, 14, 15);
            // B: same pattern from f32 weights, 4x float4 loads + convert
            v4f w0 = *(const v4f*)(wrow + k0);
            v4f w1 = *(const v4f*)(wrow + k0 + 4);
            v4f w2 = *(const v4f*)(wrow + k0 + 16);
            v4f w3 = *(const v4f*)(wrow + k0 + 20);
            v16h b;
#pragma unroll
            for (int j = 0; j < 4; ++j) {
                b[j]      = (_Float16)w0[j];
                b[4 + j]  = (_Float16)w1[j];
                b[8 + j]  = (_Float16)w2[j];
                b[12 + j] = (_Float16)w3[j];
            }
            acc = __builtin_amdgcn_wmma_f32_16x16x32_f16(
                false, a, false, b, (short)0, acc, false, false);
        }
        const float b1 = fc1_b[nh];
        const int mrow = (lane >> 4) << 3;
#pragma unroll
        for (int r = 0; r < 8; ++r) {
            float v = acc[r] + b1;
            hid_s[(mrow + r) * 64 + nh] = (_Float16)(v > 0.0f ? v : 0.0f);
        }
    }
    __syncthreads();

    // --- FC2: wave 0 only, 16x16 logits, K=64 ------------------------------
    if (tid < 32) {
        const int ne = lane & 15;
        const float* wrow = fc2_w + (size_t)ne * 64;
        v8f acc = {};
#pragma unroll
        for (int s = 0; s < 2; ++s) {
            const int k0 = s * 32 + kb;
            v8h alo = *(const v8h*)(hid_s + (lane & 15) * 64 + k0);
            v8h ahi = *(const v8h*)(hid_s + (lane & 15) * 64 + k0 + 16);
            v16h a = __builtin_shufflevector(alo, ahi,
                0, 1, 2, 3, 4, 5, 6, 7, 8, 9, 10, 11, 12, 13, 14, 15);
            v4f w0 = *(const v4f*)(wrow + k0);
            v4f w1 = *(const v4f*)(wrow + k0 + 4);
            v4f w2 = *(const v4f*)(wrow + k0 + 16);
            v4f w3 = *(const v4f*)(wrow + k0 + 20);
            v16h b;
#pragma unroll
            for (int j = 0; j < 4; ++j) {
                b[j]      = (_Float16)w0[j];
                b[4 + j]  = (_Float16)w1[j];
                b[8 + j]  = (_Float16)w2[j];
                b[12 + j] = (_Float16)w3[j];
            }
            acc = __builtin_amdgcn_wmma_f32_16x16x32_f16(
                false, a, false, b, (short)0, acc, false, false);
        }
        float tc = temp[0];
        tc = fminf(fmaxf(tc, 0.5f), 5.0f);
        const float inv_t = 1.0f / tc;
        const float b2 = fc2_b[ne];
        const int mrow = (lane >> 4) << 3;
#pragma unroll
        for (int r = 0; r < 8; ++r)
            logit_s[(mrow + r) * 16 + ne] = (acc[r] + b2) * inv_t;
    }
    __syncthreads();

    // --- top-2, softmax over the 2, scatter gates --------------------------
    if (tid < 16) {
        const int row = tid;
        float l[16];
#pragma unroll
        for (int j = 0; j < 16; ++j) l[j] = logit_s[row * 16 + j];
        int i1 = 0; float v1 = l[0];
#pragma unroll
        for (int j = 1; j < 16; ++j)
            if (l[j] > v1) { v1 = l[j]; i1 = j; }
        int i2 = -1; float v2 = -INFINITY;
#pragma unroll
        for (int j = 0; j < 16; ++j)
            if (j != i1 && l[j] > v2) { v2 = l[j]; i2 = j; }

        float e2 = __expf(v2 - v1);           // stable 2-way softmax
        float sm = 1.0f + e2;
        float g1 = 1.0f / sm, g2 = e2 / sm;
        float denom = g1 + g2 + 1e-10f;
        g1 /= denom; g2 /= denom;

        const int gb = row0 + row;
        float* grow = gates + (size_t)gb * 16;
#pragma unroll
        for (int j = 0; j < 16; ++j) grow[j] = 0.0f;
        grow[i1] = g1;
        grow[i2] = g2;
        idx_out[gb * 2 + 0] = i1;
        idx_out[gb * 2 + 1] = i2;
        float* lrow = logits_out + (size_t)gb * 16;
#pragma unroll
        for (int j = 0; j < 16; ++j) lrow[j] = l[j];
    }
}

// ---------------------------------------------------------------------------
extern "C" void kernel_launch(void* const* d_in, const int* in_sizes, int n_in,
                              void* d_out, int out_size, void* d_ws, size_t ws_size,
                              hipStream_t stream) {
    const float* x      = (const float*)d_in[0];
    const float* conv_w = (const float*)d_in[1];
    const float* conv_b = (const float*)d_in[2];
    const float* fc1_w  = (const float*)d_in[3];
    const float* fc1_b  = (const float*)d_in[4];
    const float* fc2_w  = (const float*)d_in[5];
    const float* fc2_b  = (const float*)d_in[6];
    const float* temp   = (const float*)d_in[7];
    // d_in[8] = noise_scale: unused (eval mode adds no noise)

    _Float16* feats = (_Float16*)d_ws;      // [256,256] f16 = 128 KB scratch

    float* out    = (float*)d_out;
    float* gates  = out;                     // [256,16]
    int*   idxo   = (int*)(out + 256 * 16);  // [256,2] int32
    float* logits = out + 256 * 16 + 256 * 2;// [256,16]

    dim3 g1(4, 256);                         // (row-bin, image)
    conv_pool_kernel<<<g1, 256, 0, stream>>>(x, conv_w, conv_b, feats);
    fc_gate_kernel<<<16, 128, 0, stream>>>(feats, fc1_w, fc1_b, fc2_w, fc2_b,
                                           temp, gates, idxo, logits);
}
